// StandardHeteroLayer_1099511628146
// MI455X (gfx1250) — matrix operand order repaired and checked by reading, not compiled, not run on hardware
//
#include <hip/hip_runtime.h>
#include <hip/hip_bf16.h>

// ---------------------------------------------------------------------------
// Types for CDNA5 WMMA (wave32, 16x16x32 bf16 -> f32)
// ---------------------------------------------------------------------------
typedef __attribute__((ext_vector_type(16))) __bf16   v16bf;
typedef __attribute__((ext_vector_type(2)))  __bf16   v2bf;
typedef __attribute__((ext_vector_type(8)))  float    v8f;
typedef __attribute__((ext_vector_type(8)))  unsigned v8u;

#define H 128
#define BM 64           // output rows per block (all M here are multiples of 64)
#define BK 32           // K chunk (bf16 WMMA K)
#define LST 17          // LDS row stride in dwords (16 k-pair dwords + 1 pad)
                        // gcd(17,64)=1 -> conflict-free across 16-lane groups

// pack two fp32 into one dword of two bf16 (lowers to v_cvt_pk_bf16_f32)
__device__ __forceinline__ unsigned pk2(float lo, float hi)
{
    v2bf v; v[0] = (__bf16)lo; v[1] = (__bf16)hi;
    return __builtin_bit_cast(unsigned, v);
}

// ---------------------------------------------------------------------------
// Segmented GEMM: C[M x 128] = epilog( sum_s A_s[M x 128] @ W[s*128.., :128] + b )
// bf16 WMMA, fp32 accumulation, double-buffered LDS software pipeline.
// LDS is dword-granular: ldsA[row][k/2], ldsWt[n][k/2] (K-pairs packed), so
// staging is packed b32 stores and fragment builds are pure dword loads that
// match the ISA 7.12.2 VGPR layouts directly.
// MASK: rows with cnt[m]==0 forced to 0 (scatter_mean empty-bucket semantics
// when the linear is applied after aggregation). RELU: max(0,x).
// In-place (C == A_s) is safe: each block writes only rows it read, at the end.
// ---------------------------------------------------------------------------
template<int NSEG, bool MASK, bool RELU>
__global__ __launch_bounds__(256)
void gemm_seg_bf16_kernel(const float* __restrict__ A0,
                          const float* __restrict__ A1,
                          const float* __restrict__ A2,
                          const float* __restrict__ W,   // [NSEG*128, 128] row-major
                          const float* __restrict__ bias,// [128]
                          const int*   __restrict__ cnt, // row mask (MASK only)
                          float* __restrict__ C,         // [M, 128]
                          int M)
{
    __shared__ unsigned ldsA[2][BM * LST];   // [64 rows][16 k-pair dwords]
    __shared__ unsigned ldsWt[2][H * LST];   // [128 n ][16 k-pair dwords]

    const int tid  = threadIdx.x;
    const int lane = tid & 31;
    const int wv   = tid >> 5;          // 0..7
    const int wm   = wv & 1;            // 2 M-groups of 32 rows
    const int wn   = wv >> 1;           // 4 N-groups of 32 cols
    const int half = lane >> 4;         // lanes 16..31 -> 1
    const int l16  = lane & 15;

    const int bm = blockIdx.x * BM;

    // staging coordinates (fixed per thread)
    const int ra  = tid >> 2;           // A row 0..63 (4 threads/row)
    const int cba = (tid & 3) * 8;      // A col base (elements)
    const int cbd = (tid & 3) * 4;      // A dword base (k-pairs)
    int gr = bm + ra; if (gr >= M) gr = M - 1;
    const int pr  = tid >> 4;           // W k-pair row 0..15 (k = 2*pr)
    const int nb  = (tid & 15) * 8;     // W n base (8 n per thread)

    constexpr int NCH = NSEG * 4;       // K chunks of 32 over NSEG*128

    float4 a0, a1, w0, w1, w2, w3;      // in-flight global data
    auto load_chunk = [&](int t) {
        const int s  = t >> 2;
        const int k0 = (t & 3) * BK;
        const float* A = (s == 0) ? A0 : ((s == 1) ? A1 : A2);
        const float* sa = A + (size_t)gr * H + k0 + cba;
        a0 = ((const float4*)sa)[0];
        a1 = ((const float4*)sa)[1];
        const float* sw0 = W + (size_t)(s * H + k0 + 2 * pr) * H + nb; // row k
        const float* sw1 = sw0 + H;                                    // row k+1
        w0 = ((const float4*)sw0)[0];
        w1 = ((const float4*)sw0)[1];
        w2 = ((const float4*)sw1)[0];
        w3 = ((const float4*)sw1)[1];
    };

    load_chunk(0);
    v8f acc[2][2] = {};

#pragma unroll
    for (int t = 0; t < NCH; ++t) {
        const int buf = t & 1;

        // ---- commit in-flight chunk to LDS as packed bf16 dwords ----
        {
            unsigned* d = &ldsA[buf][ra * LST + cbd];
            d[0] = pk2(a0.x, a0.y);
            d[1] = pk2(a0.z, a0.w);
            d[2] = pk2(a1.x, a1.y);
            d[3] = pk2(a1.z, a1.w);
            // W: dword [n][pr] = (W[k][n], W[k+1][n])
            unsigned* wd = &ldsWt[buf][nb * LST + pr];
            wd[0 * LST] = pk2(w0.x, w2.x);
            wd[1 * LST] = pk2(w0.y, w2.y);
            wd[2 * LST] = pk2(w0.z, w2.z);
            wd[3 * LST] = pk2(w0.w, w2.w);
            wd[4 * LST] = pk2(w1.x, w3.x);
            wd[5 * LST] = pk2(w1.y, w3.y);
            wd[6 * LST] = pk2(w1.z, w3.z);
            wd[7 * LST] = pk2(w1.w, w3.w);
        }
        __syncthreads();                 // single barrier per chunk (double buffer)

        // ---- prefetch next chunk's global data while WMMAs run ----
        if (t + 1 < NCH) load_chunk(t + 1);

        // ---- build fragments (ISA 7.12.2 layouts) and issue WMMA ----
        v16bf afrag[2];
#pragma unroll
        for (int mi = 0; mi < 2; ++mi) {
            const unsigned* ar = &ldsA[buf][(wm * 32 + mi * 16 + l16) * LST];
            v8u u;
#pragma unroll
            for (int j = 0; j < 4; ++j) {
                u[j]     = ar[half * 4 + j];        // VGPR j  : K pairs 0..7  (+8 hi-lanes)
                u[4 + j] = ar[8 + half * 4 + j];    // VGPR 4+j: K pairs 16..23(+8 hi-lanes)
            }
            afrag[mi] = __builtin_bit_cast(v16bf, u);
        }
        v16bf bfrag[2];
#pragma unroll
        for (int ni = 0; ni < 2; ++ni) {
            const unsigned* br = &ldsWt[buf][(wn * 32 + ni * 16 + l16) * LST];
            v8u u;
#pragma unroll
            for (int j = 0; j < 4; ++j) {
                u[j]     = br[half * 4 + j];        // VGPR j  : K pairs 0..7  (+8 hi-lanes)
                u[4 + j] = br[8 + half * 4 + j];    // VGPR 4+j: K pairs 16..23(+8 hi-lanes)
            }
            bfrag[ni] = __builtin_bit_cast(v16bf, u);
        }
#pragma unroll
        for (int mi = 0; mi < 2; ++mi)
#pragma unroll
            for (int ni = 0; ni < 2; ++ni)
                acc[mi][ni] = __builtin_amdgcn_wmma_f32_16x16x32_bf16(
                    false, afrag[mi], false, bfrag[ni],
                    (short)0, acc[mi][ni], false, false);
    }

    // ---- straight-line epilogue: bias, (mask), (relu), fp32 store ----
#pragma unroll
    for (int mi = 0; mi < 2; ++mi) {
#pragma unroll
        for (int ni = 0; ni < 2; ++ni) {
            const int ncol = wn * 32 + ni * 16 + l16;
            const float bv = bias[ncol];
#pragma unroll
            for (int j = 0; j < 8; ++j) {
                const int m = bm + wm * 32 + mi * 16 + j + half * 8;
                float v = acc[mi][ni][j] + bv;
                if (MASK) v = (cnt[m] == 0) ? 0.0f : v;   // v_cndmask, no branch
                if (RELU) v = fmaxf(v, 0.0f);
                C[(size_t)m * H + ncol] = v;
            }
        }
    }
}

// ---------------------------------------------------------------------------
// Scatter helpers
// ---------------------------------------------------------------------------
__global__ void count_kernel(const int* __restrict__ dst, int* __restrict__ cnt, int E)
{
    int e = blockIdx.x * 256 + threadIdx.x;
    if (e < E)
        __hip_atomic_fetch_add(&cnt[dst[e]], 1, __ATOMIC_RELAXED, __HIP_MEMORY_SCOPE_AGENT);
}

// one wave per edge; lane handles 4 floats -> float4 gather + 4 fp32 atomics
__global__ __launch_bounds__(256)
void scatter_add_kernel(const float* __restrict__ msg,
                        const int* __restrict__ src, const int* __restrict__ dst,
                        float* __restrict__ sums, int E)
{
    int e = blockIdx.x * 8 + (threadIdx.x >> 5);
    if (e >= E) return;
    int lane = threadIdx.x & 31;
    int sN = src[e];
    int dN = dst[e];
    float4 v = *(const float4*)(msg + (size_t)sN * H + lane * 4);
    float* out = sums + (size_t)dN * H + lane * 4;
    __hip_atomic_fetch_add(out + 0, v.x, __ATOMIC_RELAXED, __HIP_MEMORY_SCOPE_AGENT);
    __hip_atomic_fetch_add(out + 1, v.y, __ATOMIC_RELAXED, __HIP_MEMORY_SCOPE_AGENT);
    __hip_atomic_fetch_add(out + 2, v.z, __ATOMIC_RELAXED, __HIP_MEMORY_SCOPE_AGENT);
    __hip_atomic_fetch_add(out + 3, v.w, __ATOMIC_RELAXED, __HIP_MEMORY_SCOPE_AGENT);
}

__global__ void normalize_kernel(float* __restrict__ sums, const int* __restrict__ cnt, int total)
{
    int i = blockIdx.x * 256 + threadIdx.x;
    if (i < total) {
        int r = i >> 7;                 // /128
        int c = cnt[r]; if (c < 1) c = 1;
        sums[i] = sums[i] / (float)c;
    }
}

// ---------------------------------------------------------------------------
// Launch
// ---------------------------------------------------------------------------
extern "C" void kernel_launch(void* const* d_in, const int* in_sizes, int n_in,
                              void* d_out, int out_size, void* d_ws, size_t ws_size,
                              hipStream_t stream)
{
    const float* hr = (const float*)d_in[0];
    const float* hp = (const float*)d_in[1];
    const float* hc = (const float*)d_in[2];
    const int* p2r = (const int*)d_in[3];
    const int* c2r = (const int*)d_in[4];
    const int* r2p = (const int*)d_in[5];
    const int* r2c = (const int*)d_in[6];
    const float* W_p2r = (const float*)d_in[7];  const float* b_p2r = (const float*)d_in[8];
    const float* W_c2r = (const float*)d_in[9];  const float* b_c2r = (const float*)d_in[10];
    const float* W_r2p = (const float*)d_in[11]; const float* b_r2p = (const float*)d_in[12];
    const float* W_r2c = (const float*)d_in[13]; const float* b_r2c = (const float*)d_in[14];
    const float* W_ur  = (const float*)d_in[15]; const float* b_ur  = (const float*)d_in[16];
    const float* W_up  = (const float*)d_in[17]; const float* b_up  = (const float*)d_in[18];
    const float* W_uc  = (const float*)d_in[19]; const float* b_uc  = (const float*)d_in[20];

    const int NR    = in_sizes[0] / H;
    const int NPROD = in_sizes[1] / H;
    const int NCUST = in_sizes[2] / H;
    const int E     = in_sizes[3] / 2;

    // ---- workspace layout (zeroed region first) ----
    float* wsf = (float*)d_ws;
    size_t o = 0;
    float* sum_p2r = wsf + o; o += (size_t)NR * H;
    float* sum_c2r = wsf + o; o += (size_t)NR * H;
    float* sum_r2p = wsf + o; o += (size_t)NPROD * H;
    float* sum_r2c = wsf + o; o += (size_t)NCUST * H;
    int* cnt_p2r = (int*)(wsf + o); o += NR;
    int* cnt_c2r = (int*)(wsf + o); o += NR;
    int* cnt_r2p = (int*)(wsf + o); o += NPROD;
    int* cnt_r2c = (int*)(wsf + o); o += NCUST;
    size_t zero_bytes = o * sizeof(float);
    float* mp = wsf + o; o += (size_t)NPROD * H;   // hp @ W_p2r + b
    float* mc = wsf + o; o += (size_t)NCUST * H;   // hc @ W_c2r + b

    float* out_r = (float*)d_out;
    float* out_p = out_r + (size_t)NR * H;
    float* out_c = out_p + (size_t)NPROD * H;

    hipMemsetAsync(d_ws, 0, zero_bytes, stream);

    dim3 blk(256);

    // 1) source-side message linears (gather commutes with the linear)
    gemm_seg_bf16_kernel<1,false,false><<<dim3(NPROD / BM), blk, 0, stream>>>(
        hp, nullptr, nullptr, W_p2r, b_p2r, nullptr, mp, NPROD);
    gemm_seg_bf16_kernel<1,false,false><<<dim3(NCUST / BM), blk, 0, stream>>>(
        hc, nullptr, nullptr, W_c2r, b_c2r, nullptr, mc, NCUST);

    // 2) degree counts
    dim3 cgrid((E + 255) / 256);
    count_kernel<<<cgrid, blk, 0, stream>>>(p2r + E, cnt_p2r, E);
    count_kernel<<<cgrid, blk, 0, stream>>>(c2r + E, cnt_c2r, E);
    count_kernel<<<cgrid, blk, 0, stream>>>(r2p + E, cnt_r2p, E);
    count_kernel<<<cgrid, blk, 0, stream>>>(r2c + E, cnt_r2c, E);

    // 3) scatter-add (one wave per edge)
    dim3 sgrid((E + 7) / 8);
    scatter_add_kernel<<<sgrid, blk, 0, stream>>>(mp, p2r, p2r + E, sum_p2r, E);
    scatter_add_kernel<<<sgrid, blk, 0, stream>>>(mc, c2r, c2r + E, sum_c2r, E);
    scatter_add_kernel<<<sgrid, blk, 0, stream>>>(hr, r2p, r2p + E, sum_r2p, E);
    scatter_add_kernel<<<sgrid, blk, 0, stream>>>(hr, r2c, r2c + E, sum_r2c, E);

    // 4) mean
    normalize_kernel<<<dim3(NR * H / 256),    blk, 0, stream>>>(sum_p2r, cnt_p2r, NR * H);
    normalize_kernel<<<dim3(NR * H / 256),    blk, 0, stream>>>(sum_c2r, cnt_c2r, NR * H);
    normalize_kernel<<<dim3(NPROD * H / 256), blk, 0, stream>>>(sum_r2p, cnt_r2p, NPROD * H);
    normalize_kernel<<<dim3(NCUST * H / 256), blk, 0, stream>>>(sum_r2c, cnt_r2c, NCUST * H);

    // 5) target-side message linears (mean commutes with the linear;
    //    cnt==0 rows forced to 0 to match empty-bucket semantics). In-place.
    gemm_seg_bf16_kernel<1,true,false><<<dim3(NPROD / BM), blk, 0, stream>>>(
        sum_r2p, nullptr, nullptr, W_r2p, b_r2p, cnt_r2p, sum_r2p, NPROD);
    gemm_seg_bf16_kernel<1,true,false><<<dim3(NCUST / BM), blk, 0, stream>>>(
        sum_r2c, nullptr, nullptr, W_r2c, b_r2c, cnt_r2c, sum_r2c, NCUST);

    // 6) node updates: segmented GEMM == concat GEMM, + bias + relu
    gemm_seg_bf16_kernel<3,false,true><<<dim3(NR / BM), blk, 0, stream>>>(
        hr, sum_p2r, sum_c2r, W_ur, b_ur, nullptr, out_r, NR);
    gemm_seg_bf16_kernel<2,false,true><<<dim3(NPROD / BM), blk, 0, stream>>>(
        hp, sum_r2p, nullptr, W_up, b_up, nullptr, out_p, NPROD);
    gemm_seg_bf16_kernel<2,false,true><<<dim3(NCUST / BM), blk, 0, stream>>>(
        hc, sum_r2c, nullptr, W_uc, b_uc, nullptr, out_c, NCUST);
}
